// FewShotSegmentor_75986561401351
// MI455X (gfx1250) — matrix operand order repaired and checked by reading, not compiled, not run on hardware
//
#include <hip/hip_runtime.h>
#include <hip/hip_bf16.h>
#include <math.h>

typedef __attribute__((ext_vector_type(16))) _Float16 v16h;
typedef __attribute__((ext_vector_type(8)))  float    v8f;

// ---------------- problem constants ----------------
constexpr int B = 4, C = 256, H = 64, W = 64;
constexpr int N  = H * W;          // 4096 pixels
constexpr int NT = N / 16;         // 256 tiles of 16 pixels
constexpr int KT = C / 32;         // 8 K-fragments of 32
constexpr int FRAG = 512;          // elements per 16x32 fragment (32 lanes x 16)
constexpr int CK  = C * 9;         // 2304 = im2col K for 3x3 conv
constexpr int CKT = CK / 32;       // 72 K-fragments
constexpr int OCT = 64 / 16;       // 4 output-channel tiles
constexpr float TAU_INV = 20.0f;   // 1/0.05
constexpr float EPS = 1e-5f;

// ---------------- workspace layout (bytes) ----------------
constexpr size_t O_MT      = 0;
constexpr size_t O_INVNQ   = O_MT      + (size_t)B * N * 4;
constexpr size_t O_INVNFG  = O_INVNQ   + (size_t)B * N * 4;
constexpr size_t O_INVNCTX = O_INVNFG  + (size_t)B * N * 4;
constexpr size_t O_GAP     = O_INVNCTX + (size_t)B * N * 4;
constexpr size_t O_G       = O_GAP     + (size_t)B * C * 4;
constexpr size_t O_SUMFG   = O_G       + (size_t)B * C * 4;
constexpr size_t O_SUMM    = O_SUMFG   + (size_t)B * C * 4;
constexpr size_t O_PS      = O_SUMM    + 256;
constexpr size_t O_PSN     = O_PS      + (size_t)B * C * 4;
constexpr size_t O_SUMFQM  = O_PSN     + (size_t)B * C * 4;
constexpr size_t O_PSUM    = O_SUMFQM  + (size_t)B * C * 4;
constexpr size_t O_SUMMC   = O_PSUM    + (size_t)B * C * 4;
constexpr size_t O_MAXFG   = O_SUMMC   + 256;
constexpr size_t O_MAXCTX  = O_MAXFG   + (size_t)B * N * 4;
constexpr size_t O_MC      = O_MAXCTX  + (size_t)B * N * 4;
constexpr size_t O_PACKW   = O_MC      + (size_t)B * N * 4;
constexpr size_t O_FQP     = O_PACKW   + (size_t)OCT * CKT * FRAG * 2;
constexpr size_t O_PACKA   = O_FQP     + (size_t)B * C * N * 2;
constexpr size_t O_PACKFG  = O_PACKA   + (size_t)B * NT * KT * FRAG * 2;
constexpr size_t O_PACKCTX = O_PACKFG  + (size_t)B * NT * KT * FRAG * 2;

// fragment index maps (CDNA5 ISA §7.12.2, 16-bit A 16x32 / B 32x16)
__device__ __forceinline__ int amapK(int s, int half) {
  return (s < 8) ? (s + half * 8) : (s + 8 + half * 8);
}
__device__ __forceinline__ int bmapK(int s, int half) {
  return (s < 8) ? (s + half * 8) : (s + 8 + half * 8);
}

// async DMA of 32 contiguous bytes global -> LDS (per-lane addresses),
// tracked with ASYNCcnt (CDNA5 ISA)
__device__ __forceinline__ void async_copy32(void* lds_dst, const void* gsrc) {
  unsigned       ldsoff = (unsigned)(size_t)lds_dst;      // low 32 bits of shared ptr = LDS offset
  unsigned long long ga = (unsigned long long)(size_t)gsrc;
  asm volatile("global_load_async_to_lds_b128 %0, %1, off\n\t"
               "global_load_async_to_lds_b128 %0, %1, off offset:16"
               :: "v"(ldsoff), "v"(ga) : "memory");
}
__device__ __forceinline__ void wait_async0() {
  asm volatile("s_wait_asynccnt 0x0" ::: "memory");
}

// ---------------- small utility kernels ----------------
__global__ void k_init(float* sumM, float* sumMc) {
  if (threadIdx.x < B) { sumM[threadIdx.x] = 0.f; sumMc[threadIdx.x] = 0.f; }
}

__global__ void k_resize(const float* __restrict__ Ms, float* __restrict__ Mt,
                         float* __restrict__ sumM) {
  int t = blockIdx.x * blockDim.x + threadIdx.x;
  if (t >= B * N) return;
  int b = t / N, p = t % N, y = p / W, x = p % W;
  const float sc = 255.0f / 63.0f;
  float sy = y * sc, sx = x * sc;
  int y0 = (int)floorf(sy), x0 = (int)floorf(sx);
  int y1 = min(y0 + 1, 255), x1 = min(x0 + 1, 255);
  float wy = sy - y0, wx = sx - x0;
  const float* m = Ms + (size_t)b * 256 * 256;
  float top = m[y0 * 256 + x0] * (1 - wx) + m[y0 * 256 + x1] * wx;
  float bot = m[y1 * 256 + x0] * (1 - wx) + m[y1 * 256 + x1] * wx;
  float v = top * (1 - wy) + bot * wy;
  Mt[t] = v;
  atomicAdd(&sumM[b], v);
}

__global__ void k_rr_support(const float* __restrict__ Fs, const float* __restrict__ Mt,
                             float* __restrict__ gap, float* __restrict__ sumFg) {
  int bc = blockIdx.x, b = bc / C;
  const float* src = Fs + (size_t)bc * N;
  const float* mt  = Mt + (size_t)b * N;
  __shared__ float r1[256], r2[256];
  float s1 = 0.f, s2 = 0.f;
  for (int p = threadIdx.x; p < N; p += 256) { float v = src[p]; s1 += v; s2 += v * mt[p]; }
  r1[threadIdx.x] = s1; r2[threadIdx.x] = s2; __syncthreads();
  for (int s = 128; s > 0; s >>= 1) {
    if (threadIdx.x < s) { r1[threadIdx.x] += r1[threadIdx.x + s]; r2[threadIdx.x] += r2[threadIdx.x + s]; }
    __syncthreads();
  }
  if (threadIdx.x == 0) { gap[bc] = r1[0] / (float)N; sumFg[bc] = r2[0]; }
}

__global__ void k_gate(const float* __restrict__ gap,
                       const float* __restrict__ w1, const float* __restrict__ b1,
                       const float* __restrict__ w2, const float* __restrict__ b2,
                       float* __restrict__ g) {
  int b = blockIdx.x;
  __shared__ float h[16];
  if (threadIdx.x < 16) {
    float a = b1[threadIdx.x];
    for (int c = 0; c < C; ++c) a += w1[threadIdx.x * C + c] * gap[b * C + c];
    h[threadIdx.x] = fminf(fmaxf(a, 0.f), 6.f);
  }
  __syncthreads();
  int c = threadIdx.x;
  float a = b2[c];
  for (int j = 0; j < 16; ++j) a += w2[c * 16 + j] * h[j];
  g[b * C + c] = 1.f / (1.f + expf(-a));
}

__global__ void k_norms(const float* __restrict__ Fs, const float* __restrict__ Fq,
                        const float* __restrict__ Mt, const float* __restrict__ g,
                        float* __restrict__ inq, float* __restrict__ infg,
                        float* __restrict__ inctx) {
  int t = blockIdx.x * blockDim.x + threadIdx.x;
  if (t >= B * N) return;
  int b = t / N, p = t % N;
  float mt = Mt[t];
  float sq = 0.f, sfg = 0.f, sctx = 0.f;
  for (int c = 0; c < C; ++c) {
    float fs = Fs[((size_t)b * C + c) * N + p];
    float fq = Fq[((size_t)b * C + c) * N + p];
    float ffg = fs * mt, fctx = fs * g[b * C + c];
    sq += fq * fq; sfg += ffg * ffg; sctx += fctx * fctx;
  }
  inq[t]   = 1.f / fmaxf(sqrtf(sq),   1e-12f);
  infg[t]  = 1.f / fmaxf(sqrtf(sfg),  1e-12f);
  inctx[t] = 1.f / fmaxf(sqrtf(sctx), 1e-12f);
}

__global__ void k_ps(const float* __restrict__ sumFg, const float* __restrict__ sumM,
                     float* __restrict__ Ps, float* __restrict__ Psn) {
  int b = blockIdx.x, c = threadIdx.x;
  float ps = sumFg[b * C + c] / (sumM[b] + EPS);
  Ps[b * C + c] = ps;
  __shared__ float r[256];
  r[c] = ps * ps; __syncthreads();
  for (int s = 128; s > 0; s >>= 1) { if (c < s) r[c] += r[c + s]; __syncthreads(); }
  float inv = 1.f / fmaxf(sqrtf(r[0]), 1e-12f);
  Psn[b * C + c] = ps * inv;
}

// pack normalized features into WMMA fragment layouts
__global__ void k_pack(const float* __restrict__ Fs, const float* __restrict__ Fq,
                       const float* __restrict__ Mt, const float* __restrict__ g,
                       const float* __restrict__ inq, const float* __restrict__ infg,
                       const float* __restrict__ inctx,
                       _Float16* __restrict__ pA, _Float16* __restrict__ pFg,
                       _Float16* __restrict__ pCtx) {
  int idx = blockIdx.x;
  int buf = idx / (B * NT);
  int rem = idx % (B * NT);
  int b = rem / NT, nt = rem % NT;
  for (int i = threadIdx.x; i < KT * FRAG; i += 256) {
    int kt = i / FRAG, r = i % FRAG;
    int lane = r / 16, s = r % 16;
    int half = lane / 16, lm = lane % 16;
    int pix = nt * 16 + lm;
    size_t dst = (((size_t)b * NT + nt) * KT + kt) * FRAG + lane * 16 + s;
    if (buf == 0) {
      int k = kt * 32 + amapK(s, half);
      pA[dst] = (_Float16)(Fq[((size_t)b * C + k) * N + pix] * inq[b * N + pix]);
    } else if (buf == 1) {
      int k = kt * 32 + bmapK(s, half);
      pFg[dst] = (_Float16)(Fs[((size_t)b * C + k) * N + pix] * Mt[b * N + pix] * infg[b * N + pix]);
    } else {
      int k = kt * 32 + bmapK(s, half);
      pCtx[dst] = (_Float16)(Fs[((size_t)b * C + k) * N + pix] * g[b * C + k] * inctx[b * N + pix]);
    }
  }
}

// fused cosine-sim GEMM + row-max.
// block: 256 (8 waves, one 16-query tile each); B tiles staged in LDS via
// async global->LDS DMA, double buffered, shared by all 8 waves.
__global__ void k_sim(const v16h* __restrict__ pA, const v16h* __restrict__ pFg,
                      const v16h* __restrict__ pCtx,
                      float* __restrict__ maxFg, float* __restrict__ maxCtx) {
  __shared__ _Float16 lds[2][KT * FRAG];            // 2 x 8 KB
  int tid = threadIdx.x;
  int lane = tid & 31, wid = tid >> 5;
  int mt = blockIdx.x * 8 + wid;
  int b = blockIdx.y;
  const v16h* Bb = (blockIdx.z == 0) ? pFg : pCtx;
  float* out = (blockIdx.z == 0) ? maxFg : maxCtx;

  // A fragments resident for the whole support loop
  v16h a[KT];
#pragma unroll
  for (int kt = 0; kt < KT; ++kt)
    a[kt] = pA[(((size_t)b * NT + mt) * KT + kt) * 32 + lane];

  // stage tile 0 (8 KB = 256 threads x 32 B, async DMA to LDS)
  async_copy32(&lds[0][tid * 16], &Bb[((size_t)b * NT) * KT * 32 + tid]);
  wait_async0();
  __syncthreads();

  float rmax[8];
#pragma unroll
  for (int r = 0; r < 8; ++r) rmax[r] = -3.4e38f;

  for (int nt = 0; nt < NT; ++nt) {
    int cur = nt & 1;
    // kick off async staging of the next tile into the other buffer;
    // the DMA overlaps this tile's WMMA chain
    if (nt + 1 < NT)
      async_copy32(&lds[cur ^ 1][tid * 16],
                   &Bb[((size_t)b * NT + nt + 1) * KT * 32 + tid]);

    // pull all 8 B fragments into registers first -> one DS wait, then pure WMMA
    v16h bv[KT];
#pragma unroll
    for (int kt = 0; kt < KT; ++kt)
      bv[kt] = *(const v16h*)&lds[cur][(size_t)kt * FRAG + lane * 16];

    // two independent accumulation chains for better matrix-pipe overlap
    v8f acc0 = {0.f, 0.f, 0.f, 0.f, 0.f, 0.f, 0.f, 0.f};
    v8f acc1 = {0.f, 0.f, 0.f, 0.f, 0.f, 0.f, 0.f, 0.f};
#pragma unroll
    for (int kt = 0; kt < 4; ++kt)
      acc0 = __builtin_amdgcn_wmma_f32_16x16x32_f16(false, a[kt], false, bv[kt],
                                                    (short)0, acc0, false, false);
#pragma unroll
    for (int kt = 4; kt < 8; ++kt)
      acc1 = __builtin_amdgcn_wmma_f32_16x16x32_f16(false, a[kt], false, bv[kt],
                                                    (short)0, acc1, false, false);
#pragma unroll
    for (int r = 0; r < 8; ++r) rmax[r] = fmaxf(rmax[r], acc0[r] + acc1[r]);

    wait_async0();       // next-tile DMA complete
    __syncthreads();     // everyone done reading cur / staging next
  }

  // deferred cross-lane max within each 16-lane half (tile columns)
#pragma unroll
  for (int r = 0; r < 8; ++r) {
#pragma unroll
    for (int m = 1; m <= 8; m <<= 1) rmax[r] = fmaxf(rmax[r], __shfl_xor(rmax[r], m, 32));
  }
  if (lane == 0 || lane == 16) {
    int rowbase = mt * 16 + (lane >> 4) * 8;
#pragma unroll
    for (int r = 0; r < 8; ++r)
      out[(size_t)b * N + rowbase + r] = rmax[r] * TAU_INV;
  }
}

__global__ void k_fuse(const float* __restrict__ mf, const float* __restrict__ mc,
                       const float* __restrict__ fw, const float* __restrict__ fb,
                       float* __restrict__ Mc, float* __restrict__ sumMc) {
  int t = blockIdx.x * blockDim.x + threadIdx.x;
  if (t >= B * N) return;
  int b = t / N, p = t % N, y = p / W, x = p % W;
  float s0 = 0.f, s1 = 0.f;
  for (int dy = -2; dy <= 2; ++dy)
    for (int dx = -2; dx <= 2; ++dx) {
      int yy = y + dy, xx = x + dx;
      if (yy >= 0 && yy < H && xx >= 0 && xx < W) {
        s0 += mf[(size_t)b * N + yy * W + xx];
        s1 += mc[(size_t)b * N + yy * W + xx];
      }
    }
  float v = 1.f / (1.f + expf(-(fw[0] * (s0 / 25.f) + fw[1] * (s1 / 25.f) + fb[0])));
  Mc[t] = v;
  atomicAdd(&sumMc[b], v);
}

__global__ void k_rr_query(const float* __restrict__ Fq, const float* __restrict__ Mc,
                           float* __restrict__ sumFqM) {
  int bc = blockIdx.x, b = bc / C;
  const float* src = Fq + (size_t)bc * N;
  const float* m = Mc + (size_t)b * N;
  __shared__ float r[256];
  float s = 0.f;
  for (int p = threadIdx.x; p < N; p += 256) s += src[p] * m[p];
  r[threadIdx.x] = s; __syncthreads();
  for (int st = 128; st > 0; st >>= 1) { if (threadIdx.x < st) r[threadIdx.x] += r[threadIdx.x + st]; __syncthreads(); }
  if (threadIdx.x == 0) sumFqM[bc] = r[0];
}

__global__ void k_pq(const float* __restrict__ sumFqM, const float* __restrict__ sumMc,
                     const float* __restrict__ Psn, float* __restrict__ psum) {
  int b = blockIdx.x, c = threadIdx.x;
  float pq = sumFqM[b * C + c] / (sumMc[b] + EPS);
  __shared__ float r[256];
  r[c] = pq * pq; __syncthreads();
  for (int s = 128; s > 0; s >>= 1) { if (c < s) r[c] += r[c + s]; __syncthreads(); }
  float inv = 1.f / fmaxf(sqrtf(r[0]), 1e-12f);
  psum[b * C + c] = pq * inv + Psn[b * C + c];
}

__global__ void k_fqprime(const float* __restrict__ Fq, const float* __restrict__ inq,
                          const float* __restrict__ psum, const float* __restrict__ Ps,
                          const float* __restrict__ gamma, _Float16* __restrict__ Fqp) {
  int t = blockIdx.x * blockDim.x + threadIdx.x;
  if (t >= B * N) return;
  int b = t / N, p = t % N;
  float S = 0.f;
  for (int c = 0; c < C; ++c) S += Fq[((size_t)b * C + c) * N + p] * psum[b * C + c];
  S *= inq[t];
  float gS = gamma[0] * S;
  for (int c = 0; c < C; ++c) {
    float v = Fq[((size_t)b * C + c) * N + p] + gS * Ps[b * C + c];
    Fqp[((size_t)b * C + c) * N + p] = (_Float16)v;
  }
}

__global__ void k_packw(const float* __restrict__ w1, _Float16* __restrict__ pW) {
  int i = blockIdx.x * blockDim.x + threadIdx.x;
  if (i >= OCT * CKT * FRAG) return;
  int mt = i / (CKT * FRAG);
  int rem = i % (CKT * FRAG);
  int kt = rem / FRAG, r = rem % FRAG;
  int lane = r / 16, s = r % 16;
  int half = lane / 16, lm = lane % 16;
  int k = kt * 32 + amapK(s, half);
  int oc = mt * 16 + lm;
  int c = k / 9, tap = k % 9;
  float v = w1[(((size_t)oc * C + c) * 3 + tap / 3) * 3 + (tap % 3)];
  pW[(((size_t)mt * CKT + kt) * FRAG) + lane * 16 + s] = (_Float16)v;
}

// decoder: 3x3 conv (256->64) as implicit WMMA GEMM + bias + ReLU + 1x1 conv (64->1)
// double-buffered on-the-fly im2col in LDS
__global__ void k_conv(const _Float16* __restrict__ Fqp, const v16h* __restrict__ pW,
                       const float* __restrict__ b1, const float* __restrict__ w2,
                       const float* __restrict__ b2, float* __restrict__ out) {
  int lane = threadIdx.x & 31, wid = threadIdx.x >> 5;
  int b = blockIdx.y;
  int tile = blockIdx.x;
  int y0 = tile / 4, x0 = (tile % 4) * 16;

  __shared__ _Float16 bfrag[2][FRAG];
  __shared__ float pred[OCT * 16];

  auto stage = [&](int buf, int kt) {
    for (int i = threadIdx.x; i < FRAG; i += 128) {
      int ln = i / 16, s = i % 16;
      int half = ln / 16, n = ln % 16;
      int k = kt * 32 + bmapK(s, half);
      int c = k / 9, tap = k % 9;
      int py = y0 + tap / 3 - 1;
      int px = x0 + n + (tap % 3) - 1;
      _Float16 v = (_Float16)0.f;
      if (py >= 0 && py < H && px >= 0 && px < W)
        v = Fqp[((size_t)b * C + c) * N + py * W + px];
      bfrag[buf][i] = v;
    }
  };

  stage(0, 0);
  __syncthreads();

  v8f acc = {0.f, 0.f, 0.f, 0.f, 0.f, 0.f, 0.f, 0.f};
  for (int kt = 0; kt < CKT; ++kt) {
    int cur = kt & 1;
    if (kt + 1 < CKT) stage(cur ^ 1, kt + 1);   // overlaps this kt's WMMA
    v16h bv = *(const v16h*)&bfrag[cur][lane * 16];
    v16h av = pW[((size_t)wid * CKT + kt) * 32 + lane];
    acc = __builtin_amdgcn_wmma_f32_16x16x32_f16(
        false, av, false, bv, (short)0, acc, false, false);
    __syncthreads();
  }

  int half = lane >> 4, n = lane & 15;
  float partial = 0.f;
#pragma unroll
  for (int r = 0; r < 8; ++r) {
    int oc = wid * 16 + r + half * 8;
    float yv = fmaxf(acc[r] + b1[oc], 0.f);
    partial += w2[oc] * yv;
  }
  partial += __shfl_down(partial, 16, 32);
  if (lane < 16) pred[wid * 16 + n] = partial;
  __syncthreads();
  if (threadIdx.x < 16) {
    float tot = b2[0];
#pragma unroll
    for (int w = 0; w < OCT; ++w) tot += pred[w * 16 + threadIdx.x];
    out[(size_t)b * N + y0 * W + x0 + threadIdx.x] = tot;
  }
}

// ---------------- host-side launch ----------------
extern "C" void kernel_launch(void* const* d_in, const int* in_sizes, int n_in,
                              void* d_out, int out_size, void* d_ws, size_t ws_size,
                              hipStream_t stream) {
  const float* Fs   = (const float*)d_in[0];
  const float* Ms   = (const float*)d_in[1];
  const float* Fq   = (const float*)d_in[2];
  const float* gw1  = (const float*)d_in[3];
  const float* gb1  = (const float*)d_in[4];
  const float* gw2  = (const float*)d_in[5];
  const float* gb2  = (const float*)d_in[6];
  const float* fw   = (const float*)d_in[7];
  const float* fb   = (const float*)d_in[8];
  const float* gamma= (const float*)d_in[9];
  const float* dw1  = (const float*)d_in[10];
  const float* db1  = (const float*)d_in[11];
  const float* dw2  = (const float*)d_in[12];
  const float* db2  = (const float*)d_in[13];
  float* out = (float*)d_out;

  char* ws = (char*)d_ws;
  float* Mt      = (float*)(ws + O_MT);
  float* inq     = (float*)(ws + O_INVNQ);
  float* infg    = (float*)(ws + O_INVNFG);
  float* inctx   = (float*)(ws + O_INVNCTX);
  float* gap     = (float*)(ws + O_GAP);
  float* g       = (float*)(ws + O_G);
  float* sumFg   = (float*)(ws + O_SUMFG);
  float* sumM    = (float*)(ws + O_SUMM);
  float* Ps      = (float*)(ws + O_PS);
  float* Psn     = (float*)(ws + O_PSN);
  float* sumFqM  = (float*)(ws + O_SUMFQM);
  float* psum    = (float*)(ws + O_PSUM);
  float* sumMc   = (float*)(ws + O_SUMMC);
  float* maxFg   = (float*)(ws + O_MAXFG);
  float* maxCtx  = (float*)(ws + O_MAXCTX);
  float* Mc      = (float*)(ws + O_MC);
  _Float16* pW   = (_Float16*)(ws + O_PACKW);
  _Float16* Fqp  = (_Float16*)(ws + O_FQP);
  _Float16* pA   = (_Float16*)(ws + O_PACKA);
  _Float16* pFg  = (_Float16*)(ws + O_PACKFG);
  _Float16* pCtx = (_Float16*)(ws + O_PACKCTX);

  k_init<<<1, 32, 0, stream>>>(sumM, sumMc);
  k_resize<<<(B * N + 255) / 256, 256, 0, stream>>>(Ms, Mt, sumM);
  k_rr_support<<<B * C, 256, 0, stream>>>(Fs, Mt, gap, sumFg);
  k_gate<<<B, 256, 0, stream>>>(gap, gw1, gb1, gw2, gb2, g);
  k_norms<<<(B * N + 255) / 256, 256, 0, stream>>>(Fs, Fq, Mt, g, inq, infg, inctx);
  k_ps<<<B, 256, 0, stream>>>(sumFg, sumM, Ps, Psn);
  k_pack<<<3 * B * NT, 256, 0, stream>>>(Fs, Fq, Mt, g, inq, infg, inctx, pA, pFg, pCtx);

  dim3 simGrid(NT / 8, B, 2);
  k_sim<<<simGrid, 256, 0, stream>>>((const v16h*)pA, (const v16h*)pFg,
                                     (const v16h*)pCtx, maxFg, maxCtx);

  k_fuse<<<(B * N + 255) / 256, 256, 0, stream>>>(maxFg, maxCtx, fw, fb, Mc, sumMc);
  k_rr_query<<<B * C, 256, 0, stream>>>(Fq, Mc, sumFqM);
  k_pq<<<B, 256, 0, stream>>>(sumFqM, sumMc, Psn, psum);
  k_fqprime<<<(B * N + 255) / 256, 256, 0, stream>>>(Fq, inq, psum, Ps, gamma, Fqp);
  k_packw<<<(OCT * CKT * FRAG + 255) / 256, 256, 0, stream>>>(dw1, pW);

  dim3 convGrid(NT, B);
  k_conv<<<convGrid, 128, 0, stream>>>(Fqp, (const v16h*)pW, db1, dw2, db2, out);
}
